// GraphCNN_87866440942328
// MI455X (gfx1250) — matrix operand order repaired and checked by reading, not compiled, not run on hardware
//
#include <hip/hip_runtime.h>

#define DD 128   // hidden dim (matches reference)

typedef _Float16 v16h __attribute__((ext_vector_type(16)));
typedef float    v8f  __attribute__((ext_vector_type(8)));
typedef __attribute__((address_space(3))) float lds_float;

// ---------------------------------------------------------------------------
// WMMA GEMM:  C[N,128] = op(A)[N,128] @ W[128,128] + bias, W given transposed
// f16 (Bt[n][k] = W[k][n]).  Optionally fuses BN(scale,shift)+ReLU on A.
// Also atomically accumulates per-output-column sum / sum-of-squares for the
// following BatchNorm (training mode, over axis 0).
//
// Block = 256 threads = 8 waves computing a 32(M) x 128(N) tile.
// A tile (32x128 f32, 16 KB) is staged once into LDS with CDNA5 async
// global->LDS copies (ASYNCcnt + s_wait_asynccnt), then each wave runs two
// 16x16 WMMA tiles sharing one B fragment: 8 x v_wmma_f32_16x16x32_f16.
// ---------------------------------------------------------------------------
__global__ __launch_bounds__(256)
void gin_gemm_wmma(const float* __restrict__ A,
                   const _Float16* __restrict__ Bt,
                   const float* __restrict__ bias,
                   const float* __restrict__ in_scale,
                   const float* __restrict__ in_shift,
                   int fuse_bn,
                   float* __restrict__ C,
                   float* __restrict__ colsum,
                   float* __restrict__ colsumsq,
                   int nrows) {
  __shared__ float tileA[32 * DD];           // 16 KB

  const int tid  = threadIdx.x;
  const int wv   = tid >> 5;                 // wave 0..7 -> output column tile
  const int lane = tid & 31;
  const int lr   = lane & 15;
  const int hi   = lane >> 4;                // 0/1: lane half
  const int hi8  = hi << 3;
  const int row0 = blockIdx.x * 32;
  if (row0 >= nrows) return;

  // ---- stage A tile into LDS via async global->LDS (ASYNCcnt) ----
  lds_float* lA = (lds_float*)tileA;
#pragma unroll
  for (int c = 0; c < 4; ++c) {
    int lin = tid + 256 * c;                 // 1024 x 16B chunks
    int r   = lin >> 5;                      // row in tile (128 f32 = 32 chunks)
    int ch  = lin & 31;
    int gr  = row0 + r;
    if (gr > nrows - 1) gr = nrows - 1;      // clamp tail (rows predicated later)
    const float* gp = A + (size_t)gr * DD + ch * 4;
    asm volatile("global_load_async_to_lds_b128 %0, %1, off"
                 :: "v"(lA + (r * DD + ch * 4)), "v"(gp)
                 : "memory");
  }
  asm volatile("s_wait_asynccnt 0x0" ::: "memory");
  __syncthreads();

  const int col = (wv << 4) + lr;            // B / C column owned by this lane
  const _Float16* brow = Bt + (size_t)col * DD;
  const float* r0 = tileA + lr * DD;         // A row, M-tile 0
  const float* r1 = tileA + (16 + lr) * DD;  // A row, M-tile 1

  v8f acc0 = {};
  v8f acc1 = {};
#pragma unroll
  for (int kk = 0; kk < DD; kk += 32) {
    const int ka = kk + (hi << 4);           // A-fragment K base (16-bit A 16x32)
    const int kb = kk + (hi << 4);           // B-fragment K base
    v16h b = *(const v16h*)(brow + kb);
    v16h a0, a1;
    if (fuse_bn) {
#pragma unroll
      for (int i = 0; i < 8; ++i) {
        const int c0 = ka + i, c1 = ka + 8 + i;
        const float sc0 = in_scale[c0], sf0 = in_shift[c0];
        const float sc1 = in_scale[c1], sf1 = in_shift[c1];
        a0[i]     = (_Float16)fmaxf(r0[c0] * sc0 + sf0, 0.0f);
        a0[8 + i] = (_Float16)fmaxf(r0[c1] * sc1 + sf1, 0.0f);
        a1[i]     = (_Float16)fmaxf(r1[c0] * sc0 + sf0, 0.0f);
        a1[8 + i] = (_Float16)fmaxf(r1[c1] * sc1 + sf1, 0.0f);
      }
    } else {
#pragma unroll
      for (int i = 0; i < 8; ++i) {
        a0[i]     = (_Float16)r0[ka + i];
        a0[8 + i] = (_Float16)r0[ka + 8 + i];
        a1[i]     = (_Float16)r1[ka + i];
        a1[8 + i] = (_Float16)r1[ka + 8 + i];
      }
    }
    acc0 = __builtin_amdgcn_wmma_f32_16x16x32_f16(false, a0, false, b,
                                                  (short)0, acc0, false, false);
    acc1 = __builtin_amdgcn_wmma_f32_16x16x32_f16(false, a1, false, b,
                                                  (short)0, acc1, false, false);
  }

  // C/D layout: lane column = col, rows = mbase + v
  const float bn = bias[col];
  float s = 0.0f, s2 = 0.0f;
  const int mb0 = row0 + hi8;
  const int mb1 = row0 + 16 + hi8;
#pragma unroll
  for (int v = 0; v < 8; ++v) {
    int m = mb0 + v;
    if (m < nrows) {
      float val = acc0[v] + bn;
      C[(size_t)m * DD + col] = val;
      s += val; s2 += val * val;
    }
  }
#pragma unroll
  for (int v = 0; v < 8; ++v) {
    int m = mb1 + v;
    if (m < nrows) {
      float val = acc1[v] + bn;
      C[(size_t)m * DD + col] = val;
      s += val; s2 += val * val;
    }
  }
  atomicAdd(&colsum[col],   s);
  atomicAdd(&colsumsq[col], s2);
}

// ---------------------------------------------------------------------------
// Small helper kernels
// ---------------------------------------------------------------------------
__global__ void gin_zero_k(float* p, int n) {
  int i = blockIdx.x * blockDim.x + threadIdx.x;
  if (i < n) p[i] = 0.0f;
}

__global__ void gin_convert_w_k(const float* __restrict__ w1,
                                const float* __restrict__ w2,
                                _Float16* __restrict__ w1t,
                                _Float16* __restrict__ w2t) {
  int idx = blockIdx.x * blockDim.x + threadIdx.x;
  if (idx >= DD * DD) return;
  int k = idx / DD, n = idx % DD;          // idx = k*DD + n
  w1t[n * DD + k] = (_Float16)w1[idx];
  w2t[n * DD + k] = (_Float16)w2[idx];
}

__global__ void gin_copy_x_k(const float* __restrict__ x, float* __restrict__ hr,
                             int hrw, int total4) {
  int idx = blockIdx.x * blockDim.x + threadIdx.x;
  if (idx >= total4) return;
  int n = idx >> 5, j = idx & 31;
  float4 v = *(const float4*)(x + (size_t)n * DD + j * 4);
  *(float4*)(hr + (size_t)n * hrw + j * 4) = v;
}

__global__ void gin_agg_init_k(const float* __restrict__ hr, int hrw, int off,
                               const float* __restrict__ eps, int l,
                               float* __restrict__ pooled, int total4) {
  int idx = blockIdx.x * blockDim.x + threadIdx.x;
  if (idx >= total4) return;
  int n = idx >> 5, j = idx & 31;
  float c = 1.0f + eps[l];
  float4 v = *(const float4*)(hr + (size_t)n * hrw + off + j * 4);
  float4 r; r.x = v.x * c; r.y = v.y * c; r.z = v.z * c; r.w = v.w * c;
  *(float4*)(pooled + (size_t)n * DD + j * 4) = r;
}

__global__ void gin_agg_edges_k(const int* __restrict__ src,
                                const int* __restrict__ dst,
                                const float* __restrict__ hr, int hrw, int off,
                                float* __restrict__ pooled, int E_) {
  long long idx = (long long)blockIdx.x * blockDim.x + threadIdx.x;
  int e = (int)(idx >> 5);
  if (e >= E_) return;
  int j = (int)(idx & 31);
  int s = src[e], d = dst[e];
  float4 v = *(const float4*)(hr + (size_t)s * hrw + off + j * 4);
  float* pp = pooled + (size_t)d * DD + j * 4;
  atomicAdd(pp + 0, v.x);
  atomicAdd(pp + 1, v.y);
  atomicAdd(pp + 2, v.z);
  atomicAdd(pp + 3, v.w);
}

__global__ void gin_bn_finalize_k(const float* __restrict__ sum,
                                  const float* __restrict__ sumsq,
                                  const float* __restrict__ g,
                                  const float* __restrict__ b,
                                  float* __restrict__ scale,
                                  float* __restrict__ shift, float invN) {
  int c = threadIdx.x;
  if (c >= DD) return;
  float mean = sum[c] * invN;
  float var  = fmaxf(sumsq[c] * invN - mean * mean, 0.0f);
  float sc   = g[c] * rsqrtf(var + 1e-5f);
  scale[c] = sc;
  shift[c] = b[c] - mean * sc;
}

__global__ void gin_bn_relu_store_k(const float* __restrict__ z,
                                    const float* __restrict__ scale,
                                    const float* __restrict__ shift,
                                    float* __restrict__ hr, int hrw, int off,
                                    int total4) {
  int idx = blockIdx.x * blockDim.x + threadIdx.x;
  if (idx >= total4) return;
  int n = idx >> 5, j = idx & 31;
  int c0 = j * 4;
  float4 v = *(const float4*)(z + (size_t)n * DD + c0);
  float4 r;
  r.x = fmaxf(v.x * scale[c0 + 0] + shift[c0 + 0], 0.0f);
  r.y = fmaxf(v.y * scale[c0 + 1] + shift[c0 + 1], 0.0f);
  r.z = fmaxf(v.z * scale[c0 + 2] + shift[c0 + 2], 0.0f);
  r.w = fmaxf(v.w * scale[c0 + 3] + shift[c0 + 3], 0.0f);
  *(float4*)(hr + (size_t)n * hrw + off + c0) = r;
}

__global__ void gin_scores_k(const float* __restrict__ hr, int hrw,
                             const float* __restrict__ aw,
                             const float* __restrict__ ab,
                             const int* __restrict__ gid,
                             float* __restrict__ score,
                             unsigned* __restrict__ smax, int N_) {
  int n = blockIdx.x * blockDim.x + threadIdx.x;
  if (n >= N_) return;
  const float* hp = hr + (size_t)n * hrw;
  float s = ab[0];
  for (int c = 0; c < hrw; ++c) s += hp[c] * aw[c];
  score[n] = s;
  unsigned bits = __float_as_uint(s);
  unsigned key  = (bits & 0x80000000u) ? ~bits : (bits | 0x80000000u);
  atomicMax(&smax[gid[n]], key);
}

__global__ void gin_expdenom_k(const float* __restrict__ score,
                               const int* __restrict__ gid,
                               const unsigned* __restrict__ smax,
                               float* __restrict__ eterm,
                               float* __restrict__ denom, int N_) {
  int n = blockIdx.x * blockDim.x + threadIdx.x;
  if (n >= N_) return;
  int g = gid[n];
  unsigned k = smax[g];
  unsigned bits = (k & 0x80000000u) ? (k & 0x7FFFFFFFu) : ~k;
  float mx = __uint_as_float(bits);
  float e = __expf(score[n] - mx);
  eterm[n] = e;
  atomicAdd(&denom[g], e);
}

__global__ void gin_gemb_k(const float* __restrict__ hr, int hrw,
                           const float* __restrict__ eterm,
                           const float* __restrict__ denom,
                           const int* __restrict__ gid,
                           float* __restrict__ gemb, int N_) {
  int idx = blockIdx.x * blockDim.x + threadIdx.x;
  int chunks = hrw >> 2;
  int n = idx / chunks;
  if (n >= N_) return;
  int j = idx % chunks;
  int g = gid[n];
  float coef = eterm[n] / denom[g];
  float4 v = *(const float4*)(hr + (size_t)n * hrw + j * 4);
  float* gp = gemb + (size_t)g * hrw + j * 4;
  atomicAdd(gp + 0, v.x * coef);
  atomicAdd(gp + 1, v.y * coef);
  atomicAdd(gp + 2, v.z * coef);
  atomicAdd(gp + 3, v.w * coef);
}

__global__ void gin_pie_k(const float* __restrict__ pi,
                          const float* __restrict__ pw,
                          const float* __restrict__ pb,
                          float* __restrict__ pie, int B_) {
  int idx = blockIdx.x * blockDim.x + threadIdx.x;
  if (idx >= B_ * 16) return;
  int b = idx / 16, j = idx % 16;
  float acc = pb[j];
  for (int k = 0; k < 25; ++k) acc += pi[b * 25 + k] * pw[k * 16 + j];
  pie[idx] = fmaxf(acc, 0.0f);
}

__global__ void gin_final_k(const float* __restrict__ gemb, int hrw,
                            const float* __restrict__ pie,
                            const float* __restrict__ ow,
                            const float* __restrict__ ob,
                            float* __restrict__ out, int B_, int OUTD_) {
  int idx = blockIdx.x * blockDim.x + threadIdx.x;
  if (idx >= B_ * OUTD_) return;
  int b = idx / OUTD_, o = idx % OUTD_;
  float acc = ob[o];
  for (int c = 0; c < hrw; ++c) acc += gemb[(size_t)b * hrw + c] * ow[c * OUTD_ + o];
  for (int k = 0; k < 16; ++k)  acc += pie[b * 16 + k] * ow[(hrw + k) * OUTD_ + o];
  out[idx] = acc;
}

// ---------------------------------------------------------------------------
static inline void* gin_carve(char*& p, size_t bytes) {
  void* r = (void*)p;
  p += (bytes + 255) & ~((size_t)255);
  return r;
}

extern "C" void kernel_launch(void* const* d_in, const int* in_sizes, int n_in,
                              void* d_out, int out_size, void* d_ws, size_t ws_size,
                              hipStream_t stream) {
  (void)n_in; (void)out_size; (void)ws_size;

  const float* x        = (const float*)d_in[0];
  const float* pi       = (const float*)d_in[1];
  const float* eps      = (const float*)d_in[2];
  const float* mlp_w1   = (const float*)d_in[3];
  const float* mlp_b1   = (const float*)d_in[4];
  const float* mlp_bn_g = (const float*)d_in[5];
  const float* mlp_bn_b = (const float*)d_in[6];
  const float* mlp_w2   = (const float*)d_in[7];
  const float* mlp_b2   = (const float*)d_in[8];
  const float* bn_g     = (const float*)d_in[9];
  const float* bn_b     = (const float*)d_in[10];
  const float* attend_w = (const float*)d_in[11];
  const float* attend_b = (const float*)d_in[12];
  const float* pi_w     = (const float*)d_in[13];
  const float* pi_b     = (const float*)d_in[14];
  const float* out_w    = (const float*)d_in[15];
  const float* out_b    = (const float*)d_in[16];
  const int*   edge_idx = (const int*)d_in[17];
  const int*   graph_id = (const int*)d_in[18];

  const int N = in_sizes[0] / DD;
  const int E = in_sizes[17] / 2;
  const int B = in_sizes[1] / 25;
  const int L = in_sizes[2];
  const int OUTD = in_sizes[16];
  const int hrw = DD * (L + 1);     // 640

  // ---- workspace carving ----
  char* p = (char*)d_ws;
  float*     hr     = (float*)    gin_carve(p, (size_t)N * hrw * 4);
  float*     pooled = (float*)    gin_carve(p, (size_t)N * DD * 4);
  float*     z1     = (float*)    gin_carve(p, (size_t)N * DD * 4);
  float*     z2     = (float*)    gin_carve(p, (size_t)N * DD * 4);
  _Float16*  w1t    = (_Float16*) gin_carve(p, (size_t)DD * DD * 2);
  _Float16*  w2t    = (_Float16*) gin_carve(p, (size_t)DD * DD * 2);
  float*     stats  = (float*)    gin_carve(p, 8 * DD * 4);
  float*     score  = (float*)    gin_carve(p, (size_t)N * 4);
  float*     eterm  = (float*)    gin_carve(p, (size_t)N * 4);
  unsigned*  smax   = (unsigned*) gin_carve(p, (size_t)B * 4);
  float*     denom  = (float*)    gin_carve(p, (size_t)B * 4);
  float*     gemb   = (float*)    gin_carve(p, (size_t)B * hrw * 4);
  float*     pie    = (float*)    gin_carve(p, (size_t)B * 16 * 4);

  const int n4       = N * (DD / 4);                 // float4 chunks per node array
  const int blk      = 256;
  const int grid_n4  = (n4 + blk - 1) / blk;
  const int grid_e   = (int)(((long long)E * 32 + blk - 1) / blk);
  const int grid_gmm = (N + 31) / 32;
  const float invN   = 1.0f / (float)N;

  // reps[0] = x
  gin_copy_x_k<<<grid_n4, blk, 0, stream>>>(x, hr, hrw, n4);

  for (int l = 0; l < L; ++l) {
    const int off = l * DD;
    float* sum1 = stats;          float* sq1 = stats + DD;
    float* sc1  = stats + 2*DD;   float* sh1 = stats + 3*DD;
    float* sum2 = stats + 4*DD;   float* sq2 = stats + 5*DD;
    float* sc2  = stats + 6*DD;   float* sh2 = stats + 7*DD;

    gin_convert_w_k<<<(DD*DD + blk - 1)/blk, blk, 0, stream>>>(
        mlp_w1 + (size_t)l*DD*DD, mlp_w2 + (size_t)l*DD*DD, w1t, w2t);

    // pooled = (1+eps)*h  ;  pooled[dst] += h[src] over edges
    gin_agg_init_k<<<grid_n4, blk, 0, stream>>>(hr, hrw, off, eps, l, pooled, n4);
    gin_agg_edges_k<<<grid_e, blk, 0, stream>>>(edge_idx, edge_idx + E,
                                                hr, hrw, off, pooled, E);

    // z1 = pooled @ W1 + b1   (+ BN stats)
    gin_zero_k<<<1, 2*DD, 0, stream>>>(sum1, 2*DD);
    gin_gemm_wmma<<<grid_gmm, blk, 0, stream>>>(pooled, w1t, mlp_b1 + l*DD,
                                                nullptr, nullptr, 0,
                                                z1, sum1, sq1, N);
    gin_bn_finalize_k<<<1, DD, 0, stream>>>(sum1, sq1, mlp_bn_g + l*DD,
                                            mlp_bn_b + l*DD, sc1, sh1, invN);

    // z2 = relu(bn(z1)) @ W2 + b2   (+ outer BN stats)
    gin_zero_k<<<1, 2*DD, 0, stream>>>(sum2, 2*DD);
    gin_gemm_wmma<<<grid_gmm, blk, 0, stream>>>(z1, w2t, mlp_b2 + l*DD,
                                                sc1, sh1, 1,
                                                z2, sum2, sq2, N);
    gin_bn_finalize_k<<<1, DD, 0, stream>>>(sum2, sq2, bn_g + l*DD,
                                            bn_b + l*DD, sc2, sh2, invN);

    // h = relu(bn(z2)) -> reps[l+1]
    gin_bn_relu_store_k<<<grid_n4, blk, 0, stream>>>(z2, sc2, sh2, hr, hrw,
                                                     off + DD, n4);
  }

  // ---- attention readout ----
  gin_zero_k<<<1, ((B + 31)/32)*32, 0, stream>>>((float*)smax, B);
  gin_zero_k<<<1, ((B + 31)/32)*32, 0, stream>>>(denom, B);
  gin_zero_k<<<(B*hrw + blk - 1)/blk, blk, 0, stream>>>(gemb, B*hrw);

  gin_scores_k<<<(N + blk - 1)/blk, blk, 0, stream>>>(hr, hrw, attend_w,
                                                      attend_b, graph_id,
                                                      score, smax, N);
  gin_expdenom_k<<<(N + blk - 1)/blk, blk, 0, stream>>>(score, graph_id, smax,
                                                        eterm, denom, N);
  gin_gemb_k<<<(N*(hrw/4) + blk - 1)/blk, blk, 0, stream>>>(hr, hrw, eterm,
                                                            denom, graph_id,
                                                            gemb, N);
  gin_pie_k<<<(B*16 + blk - 1)/blk, blk, 0, stream>>>(pi, pi_w, pi_b, pie, B);
  gin_final_k<<<(B*OUTD + blk - 1)/blk, blk, 0, stream>>>(gemb, hrw, pie,
                                                          out_w, out_b,
                                                          (float*)d_out, B, OUTD);
}
